// SpikingSSMLayer_64570538328812
// MI455X (gfx1250) — matrix operand order repaired and checked by reading, not compiled, not run on hardware
//
#include <hip/hip_runtime.h>

// ---------------------------------------------------------------------------
// Spiking SSM layer for MI455X (gfx1250, wave32, WMMA).
// Shapes fixed by the reference: B=8, T=16, S=1024, d_model=512, d_state=64.
// Memory-bound (~512MB min traffic vs ~18 GFLOP): stream x exactly once from
// HBM (nontemporal), stage the binary values as u8 in LDS for the residual,
// keep f16 weights in LDS, recurrent membranes in LDS/VGPRs, matmuls via
// v_wmma_f32_16x16x32_f16 (spikes/0-1 inputs are exact in f16).
//
// Per block (1 WGP): 4 token tiles of 16 tokens, 8 waves.
//   waves 0-3 (state):  su = h@A^T + x@B^T (72 WMMA/step), LIF, spikes->LDS,
//                       x bytes->LDS (derived from the f16 bits via v_perm).
//   waves 4-7 (output): ou = s@C^T + x + D (64 WMMA/step), LIF on vo (f16,
//                       wave-private packed LDS: 1 ds_load_b128 +
//                       1 ds_store_b128 per 16-col tile), spikes -> HBM (NT).
// One barrier per step + double-buffered spike/x staging => state wave step
// t+1 overlaps output wave step t.  All waves stay under 256 VGPRs (no
// s_set_vgpr_msb churn).
// ---------------------------------------------------------------------------

typedef __attribute__((ext_vector_type(16))) _Float16 v16h;
typedef __attribute__((ext_vector_type(8)))  _Float16 v8h;
typedef __attribute__((ext_vector_type(4)))  _Float16 v4h;
typedef __attribute__((ext_vector_type(2)))  _Float16 h2;
typedef __attribute__((ext_vector_type(2)))  __fp16   fp16x2;  // builtin's type
typedef __attribute__((ext_vector_type(8)))  float    v8f;
typedef __attribute__((ext_vector_type(4)))  float    v4f;
typedef __attribute__((ext_vector_type(2)))  unsigned v2u;

constexpr int kT  = 16;
constexpr int kS  = 1024;
constexpr int kDM = 512;   // d_model

// LDS layout (bytes). Rows padded to 16B multiples that break 64-bank strides.
constexpr int A_STR = 72;    // halfs/row for A (64x64)
constexpr int B_STR = 520;   // halfs/row for B (64x512)
constexpr int C_STR = 72;    // halfs/row for C (512x64)
constexpr int S_STR = 72;    // halfs/row for spike scratch (16 rows per tile)
constexpr int X_STR = 520;   // bytes/row for staged x (16 rows per tile)

constexpr int OFF_A  = 0;
constexpr int OFF_B  = OFF_A + 64  * A_STR * 2;             //   9216
constexpr int OFF_C  = OFF_B + 64  * B_STR * 2;             //  75776
constexpr int OFF_D  = OFF_C + 512 * C_STR * 2;             // 149504
constexpr int OFF_S  = OFF_D + 512 * 4;                     // 151552
constexpr int OFF_XS = OFF_S + 4 * 2 * 16 * S_STR * 2;      // 169984
constexpr int OFF_VO = OFF_XS + 4 * 2 * 16 * X_STR;         // 236544
constexpr int LDS_BYTES = OFF_VO + 4 * 32 * 32 * 8 * 2;     // 302080 (vo f16)
static_assert(LDS_BYTES <= 320 * 1024, "LDS over WGP budget");

__device__ __forceinline__ v8f wmma16(v16h a, v16h b, v8f c) {
  // D(16x16 f32) = A(16x32 f16) * B(32x16 f16) + C
  return __builtin_amdgcn_wmma_f32_16x16x32_f16(false, a, false, b, (short)0, c,
                                                false, false);
}

// A-operand fragment from LDS. ISA 16-bit A 16x32 layout: lanes 0-15 hold
// K = k0..k0+7 and k0+16..k0+23; lanes 16-31 are +8 (caller lane-adjusts k0).
__device__ __forceinline__ v16h frag_a_lds(const _Float16* p) {
  v8h lo = *(const v8h*)p;          // ds_load_b128
  v8h hi = *(const v8h*)(p + 16);   // ds_load_b128
  return __builtin_shufflevector(lo, hi, 0,1,2,3,4,5,6,7,8,9,10,11,12,13,14,15);
}

// B-operand fragment from LDS. 16-bit B 32x16 layout: lane n holds
// K = k0..k0+7 and k0+16..k0+23; lanes 16-31 are +8 (caller lane-adjusts the base).
__device__ __forceinline__ v16h frag_b_lds(const _Float16* p) {
  v8h lo = *(const v8h*)p;
  v8h hi = *(const v8h*)(p + 16);
  return __builtin_shufflevector(lo, hi, 0,1,2,3,4,5,6,7,8,9,10,11,12,13,14,15);
}

// v_cvt_pkrtz_f16_f32 (exact for 0/1 inputs), bit-cast to _Float16 vector.
__device__ __forceinline__ h2 cvt_pk(float xlo, float xhi) {
  fp16x2 r = __builtin_amdgcn_cvt_pkrtz(xlo, xhi);
  return __builtin_bit_cast(h2, r);
}

// Two packed halfs (each 0x0000 or 0x3C00) -> two bytes {0,1} in u32 16-bit lanes.
__device__ __forceinline__ unsigned h2bits(h2 q) {
  return (__builtin_bit_cast(unsigned, q) >> 10) & 0x00010001u;
}

__global__ void __launch_bounds__(256, 1)
snn_ssm_kernel(const float* __restrict__ x,   // (B,T,S,DM) binary
               const float* __restrict__ Aw,  // (64,64)
               const float* __restrict__ Bw,  // (64,512)
               const float* __restrict__ Cw,  // (512,64)
               const float* __restrict__ Dw,  // (512,)
               float* __restrict__ out)       // (B,T,S,DM)
{
  extern __shared__ char smem[];
  _Float16*      A_lds = (_Float16*)(smem + OFF_A);
  _Float16*      B_lds = (_Float16*)(smem + OFF_B);
  _Float16*      C_lds = (_Float16*)(smem + OFF_C);
  float*         D_lds = (float*)   (smem + OFF_D);
  _Float16*      S_lds = (_Float16*)(smem + OFF_S);
  unsigned char* X_lds = (unsigned char*)(smem + OFF_XS);
  _Float16*      VO_lds = (_Float16*)(smem + OFF_VO);

  const int tid   = threadIdx.x;
  const int lane  = tid & 31;
  const int wave  = tid >> 5;
  const int tilew = wave & 3;       // which of the 4 token tiles this wave serves
  const bool owner = wave < 4;      // waves 0-3: state path; 4-7: output path
  const int n16 = lane & 15;
  const int sel = lane >> 4;        // lane half (K-group select)

  // --- stage weights into LDS as f16 (row-padded) --------------------------
  for (int i = tid; i < 64 * 64;  i += 256) A_lds[(i >> 6) * A_STR + (i & 63)]  = (_Float16)Aw[i];
  for (int i = tid; i < 64 * 512; i += 256) B_lds[(i >> 9) * B_STR + (i & 511)] = (_Float16)Bw[i];
  for (int i = tid; i < 512 * 64; i += 256) C_lds[(i >> 6) * C_STR + (i & 63)]  = (_Float16)Cw[i];
  for (int i = tid; i < 512;      i += 256) D_lds[i] = Dw[i];

  // vo: wave-private packed layout [jo][lane][r] (f16, 16B per lane per jo)
  _Float16* vot = VO_lds + tilew * 32 * 32 * 8 + lane * 8;
  if (!owner) {
    v8h z = {};
    #pragma unroll
    for (int jo = 0; jo < 32; ++jo) *(v8h*)(vot + jo * 256) = z;
  }
  __syncthreads();

  // --- this wave's 16-token tile (contiguous s within one batch b) ---------
  const int tile = blockIdx.x * 4 + tilew;   // 512 tiles total
  const int b    = tile >> 6;                // 64 tiles per batch element
  const int s0   = (tile & 63) << 4;

  v8f  vs[4];                                // state-LIF membrane (owner)
  vs[0] = {}; vs[1] = {}; vs[2] = {}; vs[3] = {};
  v16h h0 = {}, h1 = {};                     // spikes as A-fragments (K 0..31, 32..63)

  float Dreg[32];                            // per-column bias, loaded once
  #pragma unroll
  for (int jo = 0; jo < 32; ++jo) Dreg[jo] = D_lds[jo * 16 + n16];

  for (int t = 0; t < kT; ++t) {
    const size_t base = ((size_t)(b * kT + t) * kS + s0) * kDM;
    const float* xt = x + base;
    const int phase = (tilew * 2 + (t & 1));
    _Float16*      sbuf = S_lds + phase * 16 * S_STR;  // double-buffered spikes
    unsigned char* xbuf = X_lds + phase * 16 * X_STR;  // double-buffered x bytes

    if (owner) {
      // ---- su = h @ A^T + x_t @ B^T  (M=16, N=64), f32 D-layout accs ------
      v8f su[4]; su[0] = {}; su[1] = {}; su[2] = {}; su[3] = {};

      {  // h @ A^T : batch all 8 weight fragments, then 8 back-to-back WMMAs
        v16h abf[8];
        #pragma unroll
        for (int kk = 0; kk < 2; ++kk)
          #pragma unroll
          for (int j = 0; j < 4; ++j)
            abf[kk * 4 + j] =
                frag_b_lds(A_lds + (j * 16 + n16) * A_STR + kk * 32 + sel * 8);
        #pragma unroll
        for (int kk = 0; kk < 2; ++kk)
          #pragma unroll
          for (int j = 0; j < 4; ++j)
            su[j] = wmma16(kk ? h1 : h0, abf[kk * 4 + j], su[j]);
      }

      // x_t @ B^T over K=512, staging x bytes into LDS as we go
      const float*   xrow = xt + (size_t)n16 * kDM;          // lane's token row
      const float*   pfrow = (t + 1 < kT) ? xrow + (size_t)kS * kDM : xrow;
      unsigned char* xsw  = xbuf + n16 * X_STR;
      #pragma unroll
      for (int kk = 0; kk < 16; ++kk) {
        const int k0a = kk * 32 + sel * 8;
        v4f a  = __builtin_nontemporal_load((const v4f*)(xrow + k0a));
        v4f bq = __builtin_nontemporal_load((const v4f*)(xrow + k0a + 4));
        v4f cq = __builtin_nontemporal_load((const v4f*)(xrow + k0a + 16));
        v4f dq = __builtin_nontemporal_load((const v4f*)(xrow + k0a + 20));
        __builtin_prefetch(pfrow + k0a, 0, 3);   // next step's row, no branch

        // f32 -> f16 via v_cvt_pkrtz (exact for 0/1); reuse bits for bytes
        h2 q0 = cvt_pk(a[0],  a[1]);
        h2 q1 = cvt_pk(a[2],  a[3]);
        h2 q2 = cvt_pk(bq[0], bq[1]);
        h2 q3 = cvt_pk(bq[2], bq[3]);
        h2 q4 = cvt_pk(cq[0], cq[1]);
        h2 q5 = cvt_pk(cq[2], cq[3]);
        h2 q6 = cvt_pk(dq[0], dq[1]);
        h2 q7 = cvt_pk(dq[2], dq[3]);
        v4h x0 = __builtin_shufflevector(q0, q1, 0,1,2,3);
        v4h x1 = __builtin_shufflevector(q2, q3, 0,1,2,3);
        v4h x2 = __builtin_shufflevector(q4, q5, 0,1,2,3);
        v4h x3 = __builtin_shufflevector(q6, q7, 0,1,2,3);
        v8h xlo = __builtin_shufflevector(x0, x1, 0,1,2,3,4,5,6,7);
        v8h xhi = __builtin_shufflevector(x2, x3, 0,1,2,3,4,5,6,7);
        v16h xf = __builtin_shufflevector(xlo, xhi,
                                          0,1,2,3,4,5,6,7,8,9,10,11,12,13,14,15);

        // stage binary values as bytes: one v_perm_b32 packs 4 of them
        v2u g1, g2;
        g1[0] = __builtin_amdgcn_perm(h2bits(q1), h2bits(q0), 0x06040200u);
        g1[1] = __builtin_amdgcn_perm(h2bits(q3), h2bits(q2), 0x06040200u);
        g2[0] = __builtin_amdgcn_perm(h2bits(q5), h2bits(q4), 0x06040200u);
        g2[1] = __builtin_amdgcn_perm(h2bits(q7), h2bits(q6), 0x06040200u);
        *(v2u*)(xsw + k0a)      = g1;        // ds_store_b64
        *(v2u*)(xsw + k0a + 16) = g2;

        const int k0b = kk * 32 + sel * 8;
        v16h bf[4];
        #pragma unroll
        for (int j = 0; j < 4; ++j)
          bf[j] = frag_b_lds(B_lds + (j * 16 + n16) * B_STR + k0b);
        #pragma unroll
        for (int j = 0; j < 4; ++j) su[j] = wmma16(xf, bf[j], su[j]);
      }

      // ---- LIF (tau=2, hard reset at v_th=1); spikes -> LDS (f16 exact) ---
      #pragma unroll
      for (int j = 0; j < 4; ++j) {
        #pragma unroll
        for (int r = 0; r < 8; ++r) {
          float v  = 0.5f * (vs[j][r] + su[j][r]);
          float sp = (v >= 1.0f) ? 1.0f : 0.0f;
          vs[j][r] = v - v * sp;
          sbuf[(sel * 8 + r) * S_STR + j * 16 + n16] = (_Float16)sp;
        }
      }
    }

    __syncthreads();  // publish this step's spikes + x bytes

    // spike A-fragments: owner's h for step t+1, partner's s for s @ C^T now
    {
      const _Float16* srow = sbuf + n16 * S_STR;
      h0 = frag_a_lds(srow + sel * 8);
      h1 = frag_a_lds(srow + 32 + sel * 8);
    }

    if (!owner) {
      // ---- ou = s @ C^T + x_t + D ; LIF on vo (packed f16 LDS); store -----
      float* outp = out + base;
      const unsigned char* xsr = xbuf;
      #pragma unroll
      for (int jo = 0; jo < 32; ++jo) {      // 32 n-tiles cover d_model
        const _Float16* crow = C_lds + (jo * 16 + n16) * C_STR + sel * 8;
        v16h cf0 = frag_b_lds(crow);
        v16h cf1 = frag_b_lds(crow + 32);
        v8f acc = {};
        acc = wmma16(h0, cf0, acc);
        acc = wmma16(h1, cf1, acc);

        const int   col = jo * 16 + n16;
        const float Dv  = Dreg[jo];
        v8h vop = *(v8h*)(vot + jo * 256);   // ds_load_b128, conflict-free
        v8h vnp;
        #pragma unroll
        for (int r = 0; r < 8; ++r) {
          const int   m  = sel * 8 + r;
          const float xv = (float)xsr[m * X_STR + col];   // binary residual
          const float ou = acc[r] + xv + Dv;
          float vn = 0.5f * ((float)vop[r] + ou);
          const float so = (vn >= 1.0f) ? 1.0f : 0.0f;
          vnp[r] = (_Float16)(vn - vn * so);
          __builtin_nontemporal_store(so, outp + (size_t)m * kDM + col);
        }
        *(v8h*)(vot + jo * 256) = vnp;       // ds_store_b128
      }
    }
  }
}

extern "C" void kernel_launch(void* const* d_in, const int* in_sizes, int n_in,
                              void* d_out, int out_size, void* d_ws, size_t ws_size,
                              hipStream_t stream) {
  (void)in_sizes; (void)n_in; (void)out_size; (void)d_ws; (void)ws_size;
  const float* x  = (const float*)d_in[0];
  const float* Aw = (const float*)d_in[1];
  const float* Bw = (const float*)d_in[2];
  const float* Cw = (const float*)d_in[3];
  const float* Dw = (const float*)d_in[4];
  float* out = (float*)d_out;

  (void)hipFuncSetAttribute((const void*)snn_ssm_kernel,
                            hipFuncAttributeMaxDynamicSharedMemorySize, LDS_BYTES);
  // 512 token tiles / 4 tiles per block = 128 blocks; 256 threads = 8 waves.
  snn_ssm_kernel<<<dim3(128), dim3(256), LDS_BYTES, stream>>>(x, Aw, Bw, Cw, Dw, out);
}
